// FuzzyContrastiveLearning_7370163880494
// MI455X (gfx1250) — matrix operand order repaired and checked by reading, not compiled, not run on hardware
//
#include <hip/hip_runtime.h>
#include <hip/hip_bf16.h>
#include <stdint.h>

#define N_PTS 8192
#define DIM   768
#define EPS   1e-8f

#define BM 128
#define BN 128
#define BK 32
#define NK (DIM / BK)          // 24 K-steps
#define COLS_PER_STRIP 1024

typedef __attribute__((ext_vector_type(16))) __bf16 v16bf;
typedef __attribute__((ext_vector_type(4)))  __bf16 v4bf;
typedef __attribute__((ext_vector_type(8)))  float  v8f;

__device__ __forceinline__ float bf_round(float f) {
  __bf16 b = (__bf16)f;
  return (float)b;
}

// ---------------- Stage 0: zero accumulators ----------------
__global__ void fcl_zero_kernel(float* __restrict__ p, int n) {
  int i = blockIdx.x * blockDim.x + threadIdx.x;
  if (i < n) p[i] = 0.0f;
}

// ---------------- Stage 1: bf16-consistent row norms ----------------
__global__ __launch_bounds__(256)
void fcl_norms_kernel(const float* __restrict__ x, float* __restrict__ nrm) {
  const int wave = threadIdx.x >> 5;   // 8 waves -> 8 rows per block
  const int lane = threadIdx.x & 31;
  const int row  = blockIdx.x * 8 + wave;
  const float* xr = x + (size_t)row * DIM;
  float s = 0.0f;
  for (int k = lane; k < DIM; k += 32) {
    float v = bf_round(xr[k]);   // match bf16 WMMA inputs so diag distance ~ 0
    s += v * v;
  }
#pragma unroll
  for (int m = 16; m >= 1; m >>= 1) s += __shfl_xor(s, m, 32);
  if (lane == 0) nrm[row] = s;
}

// ---------------- Stage 2: tiled bf16 WMMA Gram + fused exp/row-reduce ----------------
__global__ __launch_bounds__(256)
void fcl_gram_kernel(const float* __restrict__ x,
                     const int*   __restrict__ labels,
                     const float* __restrict__ nrm,
                     float* __restrict__ posSum,
                     float* __restrict__ allSum) {
  __shared__ uint16_t ldsA[2][BM * BK];   // 2 x 8 KB, bf16 row-major [row][k]
  __shared__ uint16_t ldsB[2][BN * BK];   // 2 x 8 KB

  const int tid  = threadIdx.x;
  const int wave = tid >> 5;           // 0..7, each wave owns 16 rows
  const int lane = tid & 31;
  const int half = lane >> 4;
  const int l16  = lane & 15;

  const int quad = tid & 7;            // which float4 within the 32-wide K slab
  const int row0 = tid >> 3;           // 0..31 (4 rows per thread, stride 32)

  const int rowBase  = blockIdx.x * BM;
  const int colStrip = blockIdx.y * COLS_PER_STRIP;

  // C/D layout: VGPR r -> row (wave*16 + r) for lanes 0-15, (+8) for lanes 16-31.
  float ni[8]; int li[8];
#pragma unroll
  for (int r = 0; r < 8; ++r) {
    int row = rowBase + wave * 16 + r + 8 * half;
    ni[r] = nrm[row];
    li[r] = labels[row];
  }

  for (int c = 0; c < COLS_PER_STRIP / BN; ++c) {
    const int colBase = colStrip + c * BN;

    float nj[8]; int lj[8];
#pragma unroll
    for (int t = 0; t < 8; ++t) {
      int col = colBase + t * 16 + l16;
      nj[t] = nrm[col];
      lj[t] = labels[col];
    }

    v8f acc[8];
#pragma unroll
    for (int t = 0; t < 8; ++t)
#pragma unroll
      for (int r = 0; r < 8; ++r) acc[t][r] = 0.0f;

    float4 stgA[4], stgB[4];

    // ---- prologue: fetch + store K-tile 0 into buffer 0 ----
#pragma unroll
    for (int rep = 0; rep < 4; ++rep) {
      const int rr = row0 + rep * 32;
      stgA[rep] = *(const float4*)(x + (size_t)(rowBase + rr) * DIM + quad * 4);
      stgB[rep] = *(const float4*)(x + (size_t)(colBase + rr) * DIM + quad * 4);
    }
#pragma unroll
    for (int rep = 0; rep < 4; ++rep) {
      const int rr = row0 + rep * 32;
      v4bf pa = { (__bf16)stgA[rep].x, (__bf16)stgA[rep].y,
                  (__bf16)stgA[rep].z, (__bf16)stgA[rep].w };
      v4bf pb = { (__bf16)stgB[rep].x, (__bf16)stgB[rep].y,
                  (__bf16)stgB[rep].z, (__bf16)stgB[rep].w };
      *(v4bf*)(&ldsA[0][rr * BK + quad * 4]) = pa;
      *(v4bf*)(&ldsB[0][rr * BK + quad * 4]) = pb;
    }
    __syncthreads();

    for (int kt = 0; kt < NK; ++kt) {
      const int cur = kt & 1;
      const int nxt = cur ^ 1;
      const bool more = (kt + 1) < NK;

      // ---- issue global fetch for next K-tile early (latency hidden by WMMAs) ----
      if (more) {
        const int k0n = (kt + 1) * BK;
#pragma unroll
        for (int rep = 0; rep < 4; ++rep) {
          const int rr = row0 + rep * 32;
          stgA[rep] = *(const float4*)(x + (size_t)(rowBase + rr) * DIM + k0n + quad * 4);
          stgB[rep] = *(const float4*)(x + (size_t)(colBase + rr) * DIM + k0n + quad * 4);
        }
      }

      // ---- preload ALL fragments (distinct regs -> pipelined ds loads) ----
      union Frag { v16bf v; uint4 q[2]; };
      Frag afrag, bfrag[8];
      {
        // A (16x32 bf16): M = l16 in both halves; K = half*8..+7 and 16+half*8..+7.
        const uint16_t* abase = &ldsA[cur][(wave * 16 + l16) * BK];
        afrag.q[0] = *(const uint4*)(abase + half * 8);
        afrag.q[1] = *(const uint4*)(abase + 16 + half * 8);
      }
#pragma unroll
      for (int t = 0; t < 8; ++t) {
        // B (32x16 bf16): lane holds column n = t*16 + l16; K = half*8..+7 and 16+half*8..+7.
        const uint16_t* bbase = &ldsB[cur][(t * 16 + l16) * BK + half * 8];
        bfrag[t].q[0] = *(const uint4*)(bbase);
        bfrag[t].q[1] = *(const uint4*)(bbase + 16);
      }

#pragma unroll
      for (int t = 0; t < 8; ++t) {
        acc[t] = __builtin_amdgcn_wmma_f32_16x16x32_bf16(
            false, afrag.v, false, bfrag[t].v, (short)0, acc[t], false, false);
      }

      // ---- convert + store next tile into the other buffer ----
      if (more) {
#pragma unroll
        for (int rep = 0; rep < 4; ++rep) {
          const int rr = row0 + rep * 32;
          v4bf pa = { (__bf16)stgA[rep].x, (__bf16)stgA[rep].y,
                      (__bf16)stgA[rep].z, (__bf16)stgA[rep].w };
          v4bf pb = { (__bf16)stgB[rep].x, (__bf16)stgB[rep].y,
                      (__bf16)stgB[rep].z, (__bf16)stgB[rep].w };
          *(v4bf*)(&ldsA[nxt][rr * BK + quad * 4]) = pa;
          *(v4bf*)(&ldsB[nxt][rr * BK + quad * 4]) = pb;
        }
      }
      __syncthreads();
    }

    // ---- epilogue: fuzzy = exp(g - (ni+nj)/2); reduce over N within the tile ----
#pragma unroll
    for (int r = 0; r < 8; ++r) {
      float rowAll = 0.0f, rowPos = 0.0f;
#pragma unroll
      for (int t = 0; t < 8; ++t) {
        float g = acc[t][r];
        float e = __expf(g - 0.5f * (ni[r] + nj[t]));
        rowAll += e;
        rowPos += (li[r] == lj[t]) ? e : 0.0f;
      }
#pragma unroll
      for (int m = 8; m >= 1; m >>= 1) {
        rowAll += __shfl_xor(rowAll, m, 32);
        rowPos += __shfl_xor(rowPos, m, 32);
      }
      if (l16 == 0) {
        int row = rowBase + wave * 16 + r + 8 * half;
        atomicAdd(&allSum[row], rowAll);
        atomicAdd(&posSum[row], rowPos);
      }
    }
  }
}

// ---------------- Stage 3: loss + mean ----------------
__global__ __launch_bounds__(256)
void fcl_finalize_kernel(const float* __restrict__ posSum,
                         const float* __restrict__ allSum,
                         float* __restrict__ out) {
  float s = 0.0f;
  for (int i = threadIdx.x; i < N_PTS; i += 256) {
    float ratio = posSum[i] / (allSum[i] + EPS);
    s += -logf(ratio);
  }
  __shared__ float red[8];
#pragma unroll
  for (int m = 16; m >= 1; m >>= 1) s += __shfl_xor(s, m, 32);
  const int wave = threadIdx.x >> 5;
  const int lane = threadIdx.x & 31;
  if (lane == 0) red[wave] = s;
  __syncthreads();
  if (threadIdx.x == 0) {
    float t = 0.0f;
#pragma unroll
    for (int w = 0; w < 8; ++w) t += red[w];
    out[0] = t / (float)N_PTS;
  }
}

extern "C" void kernel_launch(void* const* d_in, const int* in_sizes, int n_in,
                              void* d_out, int out_size, void* d_ws, size_t ws_size,
                              hipStream_t stream) {
  const float* x      = (const float*)d_in[0];
  const int*   labels = (const int*)d_in[1];   // JAX default x64-off -> int32
  float* out = (float*)d_out;

  float* nrm    = (float*)d_ws;       // [8192]
  float* posSum = nrm + N_PTS;        // [8192]
  float* allSum = posSum + N_PTS;     // [8192]  (posSum/allSum contiguous)

  fcl_zero_kernel<<<(2 * N_PTS + 255) / 256, 256, 0, stream>>>(posSum, 2 * N_PTS);
  fcl_norms_kernel<<<N_PTS / 8, 256, 0, stream>>>(x, nrm);
  dim3 grid(N_PTS / BM, N_PTS / COLS_PER_STRIP);   // 64 x 8 = 512 blocks
  fcl_gram_kernel<<<grid, 256, 0, stream>>>(x, labels, nrm, posSum, allSum);
  fcl_finalize_kernel<<<1, 256, 0, stream>>>(posSum, allSum, out);
}
